// StanceGCN_13374528160161
// MI455X (gfx1250) — compile-verified, matrix operand order checked by reading, not run on hardware
//
#include <hip/hip_runtime.h>
#include <hip/hip_bf16.h>

// ---------------------------------------------------------------------------
// StanceGCN for MI455X (gfx1250): 3 layers of
//   corr = X X^T ; attn = softmax(corr) ; trans = X W^T ; Y = relu(X + attn trans)
// All matmuls use v_wmma_f32_16x16x32_bf16 (bf16 operands, f32 accumulate).
// ---------------------------------------------------------------------------

#define NB   16          // batch
#define LT   256
#define LC   768
#define LN   1024        // LT + LC
#define FF   256         // feature dim
#define NLAYERS 3

typedef __attribute__((ext_vector_type(16))) __bf16 bf16x16;
typedef __attribute__((ext_vector_type(8)))  __bf16 bf16x8;
typedef __attribute__((ext_vector_type(8)))  float  f32x8;
typedef __attribute__((ext_vector_type(8)))  float  v8f;

// ---- d_out layout (float offsets) ----
#define OFF_TGT   0
#define OFF_CLM   (NB * LT * FF)                      // 1,048,576
#define OFF_NODE0 (OFF_CLM + NB * LC * FF)            // 4,194,304
#define NODE_SZ   (NB * LN * FF)                      // 4,194,304
#define OFF_CORR0 (OFF_NODE0 + NLAYERS * NODE_SZ)     // 16,777,216
#define CORR_SZ   (NB * LN * LN)                      // 16,777,216

// ---- workspace layout (bytes) ----
#define WS_X0     0                                   // fp32 X0: 16 MB
#define WS_XBF    (NB * LN * FF * 4)                  // bf16 X : 8 MB
#define WS_TRT    (WS_XBF + NB * LN * FF * 2)         // bf16 transT: 8 MB

// ---------------------------------------------------------------------------
// Operand loaders matching the CDNA5 16-bit A/B WMMA register layout:
// per lane (half = lane>>4): elems 0..7  = mem[row][k0 + half*8 .. +7]
//                            elems 8..15 = mem[row][k0 + 16 + half*8 .. +7]
// ---------------------------------------------------------------------------
__device__ __forceinline__ bf16x16 load_ab_bf16(const __bf16* base, int row,
                                                int stride, int k0, int half) {
  const bf16x8* p0 = (const bf16x8*)(base + (size_t)row * stride + k0 + half * 8);
  const bf16x8* p1 = (const bf16x8*)(base + (size_t)row * stride + k0 + 16 + half * 8);
  bf16x8 lo = *p0, hi = *p1;
  bf16x16 r;
#pragma unroll
  for (int i = 0; i < 8; ++i) { r[i] = lo[i]; r[i + 8] = hi[i]; }
  return r;
}

__device__ __forceinline__ bf16x16 load_ab_f32(const float* base, int row,
                                               int stride, int k0, int half) {
  const f32x8* p0 = (const f32x8*)(base + (size_t)row * stride + k0 + half * 8);
  const f32x8* p1 = (const f32x8*)(base + (size_t)row * stride + k0 + 16 + half * 8);
  f32x8 lo = *p0, hi = *p1;
  bf16x16 r;
#pragma unroll
  for (int i = 0; i < 8; ++i) { r[i] = (__bf16)lo[i]; r[i + 8] = (__bf16)hi[i]; }
  return r;
}

#define WMMA_BF16(A, B, C) \
  __builtin_amdgcn_wmma_f32_16x16x32_bf16(false, (A), false, (B), (short)0, (C), false, false)

// ---------------------------------------------------------------------------
// Kernel 0: concat(target, claim) -> X0 (fp32, ws) and Xbf (bf16, ws)
// ---------------------------------------------------------------------------
__global__ void prep_kernel(const float* __restrict__ tgt,
                            const float* __restrict__ clm,
                            float* __restrict__ X0, __bf16* __restrict__ Xbf) {
  size_t idx = (size_t)blockIdx.x * blockDim.x + threadIdx.x;  // over NB*LN*FF
  size_t f = idx & (FF - 1);
  size_t l = (idx >> 8) & (LN - 1);
  size_t b = idx >> 18;
  float v = (l < LT) ? tgt[(b * LT + l) * FF + f]
                     : clm[(b * LC + (l - LT)) * FF + f];
  X0[idx] = v;
  Xbf[idx] = (__bf16)v;
}

// ---------------------------------------------------------------------------
// Kernel 1: trans = X W^T, stored TRANSPOSED as bf16: transT[b][g][m]
// grid = (FF/16, LN/16, NB), block = 32 (one wave per 16x16 tile)
// ---------------------------------------------------------------------------
__global__ void trans_kernel(const __bf16* __restrict__ Xbf,
                             const float* __restrict__ W,
                             __bf16* __restrict__ transT) {
  int b = blockIdx.z, m0 = blockIdx.y * 16, g0 = blockIdx.x * 16;
  int lane = threadIdx.x, half = lane >> 4, ln = lane & 15;
  const __bf16* X = Xbf + (size_t)b * LN * FF;

  v8f acc = {};
#pragma unroll
  for (int k0 = 0; k0 < FF; k0 += 32) {
    bf16x16 a = load_ab_bf16(X, m0 + ln, FF, k0, half);       // rows of X
    bf16x16 bm = load_ab_f32(W, g0 + ln, FF, k0, half);       // rows of W (B cols)
    acc = WMMA_BF16(a, bm, acc);
  }
  // D: elem r -> row m0 + half*8 + r, col g0 + ln  => 8 contiguous m per lane
  bf16x8 v;
#pragma unroll
  for (int r = 0; r < 8; ++r) v[r] = (__bf16)acc[r];
  __bf16* dst = transT + (size_t)b * FF * LN + (size_t)(g0 + ln) * LN + m0 + half * 8;
  *(bf16x8*)dst = v;
}

// ---------------------------------------------------------------------------
// Kernel 2: corr row-strip + softmax -> attn written into d_out corr region.
// grid = (LN/16, NB), block = 256 (8 waves x 128 cols), dyn LDS = 16*1024*4 B
// ---------------------------------------------------------------------------
__global__ void corr_softmax_kernel(const __bf16* __restrict__ Xbf,
                                    float* __restrict__ attn_out) {
  extern __shared__ float smem[];                 // [16][1024]
  int b = blockIdx.y, m0 = blockIdx.x * 16;
  int tid = threadIdx.x, wave = tid >> 5, lane = tid & 31;
  int half = lane >> 4, ln = lane & 15;
  const __bf16* X = Xbf + (size_t)b * LN * FF;
  int n0 = wave * 128;

  v8f acc[8];
#pragma unroll
  for (int t = 0; t < 8; ++t) acc[t] = {};

#pragma unroll
  for (int k0 = 0; k0 < FF; k0 += 32) {
    bf16x16 a = load_ab_bf16(X, m0 + ln, FF, k0, half);
#pragma unroll
    for (int t = 0; t < 8; ++t) {                 // B cols = X rows (X X^T)
      bf16x16 bm = load_ab_bf16(X, n0 + t * 16 + ln, FF, k0, half);
      acc[t] = WMMA_BF16(a, bm, acc[t]);
    }
  }
  // stage 16x1024 fp32 strip in LDS
#pragma unroll
  for (int t = 0; t < 8; ++t)
#pragma unroll
    for (int r = 0; r < 8; ++r)
      smem[(r + half * 8) * LN + n0 + t * 16 + ln] = acc[t][r];
  __syncthreads();

  // row softmax: 16 threads per row
  int row = tid >> 4, j = tid & 15;
  float* srow = smem + row * LN;
  float mx = -3.402823466e38f;
  for (int c = j; c < LN; c += 16) mx = fmaxf(mx, srow[c]);
#pragma unroll
  for (int off = 8; off; off >>= 1) mx = fmaxf(mx, __shfl_xor(mx, off, 16));
  float sum = 0.f;
  for (int c = j; c < LN; c += 16) {
    float e = __expf(srow[c] - mx);
    srow[c] = e;
    sum += e;
  }
#pragma unroll
  for (int off = 8; off; off >>= 1) sum += __shfl_xor(sum, off, 16);
  float inv = 1.0f / sum;
  float* out = attn_out + ((size_t)b * LN + m0 + row) * LN;
  for (int c = j; c < LN; c += 16) out[c] = srow[c] * inv;
}

// ---------------------------------------------------------------------------
// Kernel 3: Y = relu(Xres + attn @ trans); writes Y fp32 (d_out node region)
// and Y bf16 (ws, input to next layer). grid = (LN/16, NB), block = 256.
// Wave w owns cols [w*32, w*32+32).
// ---------------------------------------------------------------------------
__global__ void av_relu_kernel(const float* __restrict__ attn,
                               const __bf16* __restrict__ transT,
                               const float* __restrict__ Xres,
                               float* __restrict__ Yout,
                               __bf16* __restrict__ Ybf) {
  int b = blockIdx.y, m0 = blockIdx.x * 16;
  int tid = threadIdx.x, wave = tid >> 5, lane = tid & 31;
  int half = lane >> 4, ln = lane & 15;
  const float* A = attn + (size_t)b * LN * LN;
  const __bf16* T = transT + (size_t)b * FF * LN;
  int n0 = wave * 32;

  v8f acc0 = {}, acc1 = {};
  for (int k0 = 0; k0 < LN; k0 += 32) {
    bf16x16 a = load_ab_f32(A, m0 + ln, LN, k0, half);            // attn rows
    bf16x16 b0 = load_ab_bf16(T, n0 + ln, LN, k0, half);          // transT rows
    bf16x16 b1 = load_ab_bf16(T, n0 + 16 + ln, LN, k0, half);
    acc0 = WMMA_BF16(a, b0, acc0);
    acc1 = WMMA_BF16(a, b1, acc1);
  }

  const float* Xr = Xres + (size_t)b * LN * FF;
  float* Y = Yout + (size_t)b * LN * FF;
  __bf16* Yb = Ybf + (size_t)b * LN * FF;
#pragma unroll
  for (int r = 0; r < 8; ++r) {
    int m = m0 + r + half * 8;
    int c0 = n0 + ln, c1 = n0 + 16 + ln;
    float y0 = fmaxf(acc0[r] + Xr[(size_t)m * FF + c0], 0.f);
    float y1 = fmaxf(acc1[r] + Xr[(size_t)m * FF + c1], 0.f);
    Y[(size_t)m * FF + c0] = y0;  Yb[(size_t)m * FF + c0] = (__bf16)y0;
    Y[(size_t)m * FF + c1] = y1;  Yb[(size_t)m * FF + c1] = (__bf16)y1;
  }
}

// ---------------------------------------------------------------------------
// Kernel 4: split final node_rep into target_rep / claim_rep
// ---------------------------------------------------------------------------
__global__ void split_copy_kernel(const float* __restrict__ node3,
                                  float* __restrict__ tgt_out,
                                  float* __restrict__ clm_out) {
  size_t idx = (size_t)blockIdx.x * blockDim.x + threadIdx.x;  // over NB*LN*FF
  size_t f = idx & (FF - 1);
  size_t l = (idx >> 8) & (LN - 1);
  size_t b = idx >> 18;
  float v = node3[idx];
  if (l < LT) tgt_out[(b * LT + l) * FF + f] = v;
  else        clm_out[(b * LC + (l - LT)) * FF + f] = v;
}

// ---------------------------------------------------------------------------
extern "C" void kernel_launch(void* const* d_in, const int* in_sizes, int n_in,
                              void* d_out, int out_size, void* d_ws, size_t ws_size,
                              hipStream_t stream) {
  const float* tgt = (const float*)d_in[0];
  const float* clm = (const float*)d_in[1];
  // d_in[2] = batch_adj: unused by the reference
  const float* W = (const float*)d_in[3];

  float* out = (float*)d_out;
  char* ws = (char*)d_ws;
  float* X0 = (float*)(ws + WS_X0);
  __bf16* Xbf = (__bf16*)(ws + WS_XBF);
  __bf16* transT = (__bf16*)(ws + WS_TRT);

  const int total = NB * LN * FF;           // 4,194,304
  prep_kernel<<<total / 256, 256, 0, stream>>>(tgt, clm, X0, Xbf);

  const float* Xf = X0;
  for (int i = 0; i < NLAYERS; ++i) {
    float* node_i = out + OFF_NODE0 + (size_t)i * NODE_SZ;
    float* corr_i = out + OFF_CORR0 + (size_t)i * CORR_SZ;

    trans_kernel<<<dim3(FF / 16, LN / 16, NB), 32, 0, stream>>>(
        Xbf, W + (size_t)i * FF * FF, transT);
    corr_softmax_kernel<<<dim3(LN / 16, NB), 256, 16 * LN * sizeof(float), stream>>>(
        Xbf, corr_i);
    av_relu_kernel<<<dim3(LN / 16, NB), 256, 0, stream>>>(
        corr_i, transT, Xf, node_i, Xbf);
    Xf = node_i;
  }

  split_copy_kernel<<<total / 256, 256, 0, stream>>>(
      out + OFF_NODE0 + (size_t)2 * NODE_SZ, out + OFF_TGT, out + OFF_CLM);
}
